// PointNet2UnetForFlowMatching_523986010542
// MI455X (gfx1250) — compile-verified, matrix operand order checked by reading, not run on hardware
//
#include <hip/hip_runtime.h>

// ---------------------------------------------------------------------------
// PointNet++ U-Net forward for MI455X (gfx1250, wave32, WMMA).
// 1x1-conv layers run as v_wmma_f32_16x16x32_f16 GEMMs: unguarded full K
// tiles, b128 global loads, column-major LDS staging so the B operand is
// read with two ds_load_b128 per WMMA pair. fp32 in HBM, fp16 in-register.
// ---------------------------------------------------------------------------

typedef __attribute__((ext_vector_type(16))) _Float16 v16h;
typedef __attribute__((ext_vector_type(8)))  _Float16 v8h;
typedef __attribute__((ext_vector_type(4)))  _Float16 v4h;
typedef __attribute__((ext_vector_type(8)))  float    v8f;
typedef __attribute__((ext_vector_type(4)))  float    v4f;

__device__ __forceinline__ float geluf(float x) {
    return 0.5f * x * (1.0f + erff(x * 0.70710678118654752f));
}

static inline int cdiv(long a, long b) { return (int)((a + b - 1) / b); }

// ---------------------------------------------------------------------------
// Time embedding: g2[b][o] = gelu(gelu(sincos(t[b]) @ time_W^T + time_b))[o]
// ---------------------------------------------------------------------------
__global__ void k_temb(const float* __restrict__ t, const float* __restrict__ Wt,
                       const float* __restrict__ bt, float* __restrict__ g2) {
    __shared__ float emb[128];
    int tid = threadIdx.x; // 128 threads
    for (int b = 0; b < 16; ++b) {
        float tv = t[b];
        if (tid < 64) {
            float f = expf((float)tid * (-logf(10000.0f) / 63.0f));
            emb[tid] = sinf(tv * f);
        } else {
            float f = expf((float)(tid - 64) * (-logf(10000.0f) / 63.0f));
            emb[tid] = cosf(tv * f);
        }
        __syncthreads();
        float s = bt[tid];
        for (int j = 0; j < 128; ++j) s += emb[j] * Wt[tid * 128 + j];
        g2[b * 128 + tid] = geluf(geluf(s));
        __syncthreads();
    }
}

// xyz (B,N,3) from x_t (B,3,N)
__global__ void k_xpose(const float* __restrict__ x, float* __restrict__ xyz, int N) {
    int i = blockIdx.x * blockDim.x + threadIdx.x;
    if (i >= 16 * N) return;
    int b = i / N, n = i % N;
    for (int c = 0; c < 3; ++c)
        xyz[(size_t)i * 3 + c] = x[((size_t)b * 3 + c) * N + n];
}

// ---------------------------------------------------------------------------
// Farthest point sampling: one block per batch; distances live in registers.
// ---------------------------------------------------------------------------
__global__ void k_fps(const float* __restrict__ xyz, int* __restrict__ out, int N, int S) {
    int b = blockIdx.x, tid = threadIdx.x; // 256 threads
    const float* P = xyz + (size_t)b * N * 3;
    float dl[32];
    int cnt0 = 0;
    for (int i = tid; i < N; i += 256) dl[cnt0++] = 1e10f;
    __shared__ float rv[256];
    __shared__ int   ri[256];
    __shared__ int   sfar;
    if (tid == 0) sfar = 0;
    __syncthreads();
    for (int s = 0; s < S; ++s) {
        int far = sfar;
        if (tid == 0) out[b * S + s] = far;
        float cx = P[far * 3 + 0], cy = P[far * 3 + 1], cz = P[far * 3 + 2];
        float bv = -1.0f; int bi = 0; int c = 0;
        for (int i = tid; i < N; i += 256, ++c) {
            float dx = P[i * 3 + 0] - cx, dy = P[i * 3 + 1] - cy, dz = P[i * 3 + 2] - cz;
            float d  = dx * dx + dy * dy + dz * dz;
            float nd = fminf(dl[c], d);
            dl[c] = nd;
            if (nd > bv) { bv = nd; bi = i; }
        }
        rv[tid] = bv; ri[tid] = bi;
        __syncthreads();
        for (int off = 128; off > 0; off >>= 1) {
            if (tid < off) {
                if (rv[tid + off] > rv[tid] ||
                    (rv[tid + off] == rv[tid] && ri[tid + off] < ri[tid])) {
                    rv[tid] = rv[tid + off]; ri[tid] = ri[tid + off];
                }
            }
            __syncthreads();
        }
        if (tid == 0) sfar = ri[0];
        __syncthreads();
    }
}

__global__ void k_gather3(const float* __restrict__ xyz, const int* __restrict__ idx,
                          float* __restrict__ out, int N, int S) {
    int i = blockIdx.x * blockDim.x + threadIdx.x;
    if (i >= 16 * S) return;
    int b = i / S;
    int gi = idx[i];
    if (gi < 0) gi = 0; if (gi >= N) gi = N - 1;
    for (int c = 0; c < 3; ++c)
        out[(size_t)i * 3 + c] = xyz[((size_t)b * N + gi) * 3 + c];
}

// Ball query: ascending scan, pad with first hit (matches sort+where semantics).
__global__ void k_ball(const float* __restrict__ xyz, const float* __restrict__ nxyz,
                       int* __restrict__ out, int N, int S, int K, float r2) {
    int i = blockIdx.x * blockDim.x + threadIdx.x;
    if (i >= 16 * S) return;
    int b = i / S;
    const float* P = xyz + (size_t)b * N * 3;
    const float* q = nxyz + (size_t)i * 3;
    float qx = q[0], qy = q[1], qz = q[2];
    int* o = out + (size_t)i * K;
    int cnt = 0, first = 0;
    for (int n = 0; n < N && cnt < K; ++n) {
        float dx = P[n * 3 + 0] - qx, dy = P[n * 3 + 1] - qy, dz = P[n * 3 + 2] - qz;
        if (dx * dx + dy * dy + dz * dz <= r2) {
            if (cnt == 0) first = n;
            o[cnt++] = n;
        }
    }
    while (cnt < K) o[cnt++] = first;
}

// Grouped SA input: X (B, 3+Cp, S*K)
__global__ void k_group(const float* __restrict__ xyz, const float* __restrict__ nxyz,
                        const float* __restrict__ pts, const int* __restrict__ gidx,
                        float* __restrict__ X, int N, int S, int K, int Cp) {
    int C = 3 + Cp;
    long i = (long)blockIdx.x * blockDim.x + threadIdx.x;
    long total = (long)16 * C * S * K;
    if (i >= total) return;
    int b  = (int)(i / ((long)C * S * K));
    int ch = (int)((i / ((long)S * K)) % C);
    int sk = (int)(i % ((long)S * K));
    int s = sk / K, k = sk % K;
    int gi = gidx[((size_t)b * S + s) * K + k];
    if (gi < 0) gi = 0; if (gi >= N) gi = N - 1;
    float v;
    if (ch < 3) v = xyz[((size_t)b * N + gi) * 3 + ch] - nxyz[((size_t)b * S + s) * 3 + ch];
    else        v = pts[((size_t)b * Cp + (ch - 3)) * N + gi];
    X[i] = v;
}

// ---------------------------------------------------------------------------
// cond_conv pieces
// tb[b][c] = bt[c] + g2[b] . Wt[c]
__global__ void k_tb(const float* __restrict__ g2, const float* __restrict__ Wt,
                     const float* __restrict__ bt, float* __restrict__ tb, int cin) {
    int i = blockIdx.x * blockDim.x + threadIdx.x;
    if (i >= 16 * cin) return;
    int b = i / cin, c = i % cin;
    float s = bt[c];
    for (int j = 0; j < 128; ++j) s += g2[b * 128 + j] * Wt[c * 128 + j];
    tb[i] = s;
}

// ofs[b][o] = bias[o] + W[o] . tb[b]
__global__ void k_ofs(const float* __restrict__ W, const float* __restrict__ bias,
                      const float* __restrict__ tb, float* __restrict__ ofs,
                      int cin, int cout) {
    int i = blockIdx.x * blockDim.x + threadIdx.x;
    if (i >= 16 * cout) return;
    int b = i / cout, o = i % cout;
    float s = bias[o];
    const float* w = W + (size_t)o * cin;
    const float* t = tb + (size_t)b * cin;
    for (int c = 0; c < cin; ++c) s += w[c] * t[c];
    ofs[i] = s;
}

__global__ void k_biasofs(const float* __restrict__ bias, float* __restrict__ ofs, int cout) {
    int i = blockIdx.x * blockDim.x + threadIdx.x;
    if (i >= 16 * cout) return;
    ofs[i] = bias[i % cout];
}

// ---------------------------------------------------------------------------
// WMMA GEMM: Y[b][o][l] = ofs[b][o] + sum_c W[o][c] * X[b][c][l]
// One wave per 32(M)x16(N) output tile (two WMMAs share one staged B tile).
// LDS tile stored column-major so each lane's B operand is 2x ds_load_b128.
// VEC4: W rows loaded as float4 (requires cin % 4 == 0).
// ---------------------------------------------------------------------------
__device__ __forceinline__ void pack4(v16h& a, int base, v4f w) {
    a[base + 0] = (_Float16)w.x; a[base + 1] = (_Float16)w.y;
    a[base + 2] = (_Float16)w.z; a[base + 3] = (_Float16)w.w;
}

template <bool VEC4>
__global__ void k_gemm_t(const float* __restrict__ W, const float* __restrict__ X,
                         const float* __restrict__ ofs, float* __restrict__ Y,
                         int cout, int cin, int L) {
    __shared__ _Float16 XsT[16 * 32]; // [col][row], col-major staging
    const int lane = threadIdx.x;     // block = 32 = one wave
    const int m = lane & 15, h = lane >> 4;
    const int b  = blockIdx.z;
    const int l0 = blockIdx.x << 4;
    const int o0 = blockIdx.y << 5;   // 32 output rows per wave
    const int orow0 = o0 + m;
    const int orow1 = o0 + 16 + m;
    const bool av0 = orow0 < cout, av1 = orow1 < cout;

    const float* xp  = X + (size_t)b * cin * L + l0;           // tile col base
    const float* wp0 = W + (size_t)(av0 ? orow0 : 0) * cin;
    const float* wp1 = W + (size_t)(av1 ? orow1 : 0) * cin;
    const int rq = lane >> 2;         // 0..7
    const int c4 = (lane & 3) << 2;   // 0,4,8,12

    v8f acc0 = {}, acc1 = {};
    const int cinFull = cin & ~31;

    for (int k0 = 0; k0 < cinFull; k0 += 32) {
        // ---- stage 32(K)x16(N) X tile: b128 loads, transposed b16 stores ----
        v4f x0 = *(const v4f*)(xp + (k0 + rq +  0) * L + c4);
        v4f x1 = *(const v4f*)(xp + (k0 + rq +  8) * L + c4);
        v4f x2 = *(const v4f*)(xp + (k0 + rq + 16) * L + c4);
        v4f x3 = *(const v4f*)(xp + (k0 + rq + 24) * L + c4);
        if (k0 + 32 < cinFull)
            __builtin_prefetch(xp + (k0 + 32 + rq) * L + c4, 0, 1); // global_prefetch_b8
        v4h h0 = { (_Float16)x0.x, (_Float16)x0.y, (_Float16)x0.z, (_Float16)x0.w };
        v4h h1 = { (_Float16)x1.x, (_Float16)x1.y, (_Float16)x1.z, (_Float16)x1.w };
        v4h h2 = { (_Float16)x2.x, (_Float16)x2.y, (_Float16)x2.z, (_Float16)x2.w };
        v4h h3 = { (_Float16)x3.x, (_Float16)x3.y, (_Float16)x3.z, (_Float16)x3.w };
#pragma unroll
        for (int j = 0; j < 4; ++j) {
            XsT[(c4 + j) * 32 + rq +  0] = h0[j];
            XsT[(c4 + j) * 32 + rq +  8] = h1[j];
            XsT[(c4 + j) * 32 + rq + 16] = h2[j];
            XsT[(c4 + j) * 32 + rq + 24] = h3[j];
        }
        __syncthreads();

        // ---- B operand: lane = col m, K pattern {h*8..+7, 16+h*8..+7} ----
        v8h blo = *(const v8h*)&XsT[m * 32 + (h << 3)];
        v8h bhi = *(const v8h*)&XsT[m * 32 + 16 + (h << 3)];
        v16h bm = __builtin_shufflevector(blo, bhi, 0, 1, 2, 3, 4, 5, 6, 7,
                                          8, 9, 10, 11, 12, 13, 14, 15);

        // ---- A operands (two 16-row tiles) ----
        v16h a0, a1;
        if (VEC4) {
            pack4(a0, 0,  *(const v4f*)(wp0 + k0 + (h << 3) + 0));
            pack4(a0, 4,  *(const v4f*)(wp0 + k0 + (h << 3) + 4));
            pack4(a0, 8,  *(const v4f*)(wp0 + k0 + 16 + (h << 3) + 0));
            pack4(a0, 12, *(const v4f*)(wp0 + k0 + 16 + (h << 3) + 4));
            pack4(a1, 0,  *(const v4f*)(wp1 + k0 + (h << 3) + 0));
            pack4(a1, 4,  *(const v4f*)(wp1 + k0 + (h << 3) + 4));
            pack4(a1, 8,  *(const v4f*)(wp1 + k0 + 16 + (h << 3) + 0));
            pack4(a1, 12, *(const v4f*)(wp1 + k0 + 16 + (h << 3) + 4));
        } else {
#pragma unroll
            for (int e = 0; e < 16; ++e) {
                int kk = ((e & 8) << 1) + (h << 3) + (e & 7);
                a0[e] = (_Float16)wp0[k0 + kk];
                a1[e] = (_Float16)wp1[k0 + kk];
            }
        }
        acc0 = __builtin_amdgcn_wmma_f32_16x16x32_f16(false, a0, false, bm,
                                                      (short)0, acc0, false, false);
        acc1 = __builtin_amdgcn_wmma_f32_16x16x32_f16(false, a1, false, bm,
                                                      (short)0, acc1, false, false);
        __syncthreads();
    }

    if (cinFull < cin) { // guarded remainder tile (cin in {6,131,259})
        const int k0 = cinFull;
#pragma unroll
        for (int rr = 0; rr < 16; ++rr) {
            int row = (rr << 1) + h;
            int K = k0 + row;
            float v = (K < cin) ? xp[K * L + m] : 0.0f;
            XsT[m * 32 + row] = (_Float16)v;
        }
        __syncthreads();
        v8h blo = *(const v8h*)&XsT[m * 32 + (h << 3)];
        v8h bhi = *(const v8h*)&XsT[m * 32 + 16 + (h << 3)];
        v16h bm = __builtin_shufflevector(blo, bhi, 0, 1, 2, 3, 4, 5, 6, 7,
                                          8, 9, 10, 11, 12, 13, 14, 15);
        v16h a0, a1;
#pragma unroll
        for (int e = 0; e < 16; ++e) {
            int kk = ((e & 8) << 1) + (h << 3) + (e & 7);
            int K = k0 + kk;
            float w0 = (K < cin) ? wp0[K] : 0.0f;
            float w1 = (K < cin) ? wp1[K] : 0.0f;
            a0[e] = (_Float16)w0;
            a1[e] = (_Float16)w1;
        }
        acc0 = __builtin_amdgcn_wmma_f32_16x16x32_f16(false, a0, false, bm,
                                                      (short)0, acc0, false, false);
        acc1 = __builtin_amdgcn_wmma_f32_16x16x32_f16(false, a1, false, bm,
                                                      (short)0, acc1, false, false);
    }

    const int col = l0 + m;
#pragma unroll
    for (int r = 0; r < 8; ++r) {
        int o = o0 + (h << 3) + r;   // D: row = 8*half + vgpr
        if (o < cout)
            Y[((size_t)b * cout + o) * L + col] = acc0[r] + ofs[b * cout + o];
        int o2 = o + 16;
        if (o2 < cout)
            Y[((size_t)b * cout + o2) * L + col] = acc1[r] + ofs[b * cout + o2];
    }
}

// Per-channel batch stats over (B, L)
__global__ void k_bnstats(const float* __restrict__ Y, float* __restrict__ mean,
                          float* __restrict__ rinv, int cout, int L) {
    int o = blockIdx.x, tid = threadIdx.x; // 256 threads
    float s = 0.f, s2 = 0.f;
    for (int b = 0; b < 16; ++b) {
        const float* p = Y + ((size_t)b * cout + o) * L;
        for (int l = tid; l < L; l += 256) { float v = p[l]; s += v; s2 += v * v; }
    }
    __shared__ float sh[256], sh2[256];
    sh[tid] = s; sh2[tid] = s2;
    __syncthreads();
    for (int off = 128; off > 0; off >>= 1) {
        if (tid < off) { sh[tid] += sh[tid + off]; sh2[tid] += sh2[tid + off]; }
        __syncthreads();
    }
    if (tid == 0) {
        float n = 16.0f * (float)L;
        float mu = sh[0] / n;
        float var = sh2[0] / n - mu * mu;
        mean[o] = mu;
        rinv[o] = rsqrtf(fmaxf(var, 0.0f) + 1e-5f);
    }
}

__global__ void k_bn_gelu(float* __restrict__ Y, const float* __restrict__ mean,
                          const float* __restrict__ rinv, const float* __restrict__ g,
                          const float* __restrict__ be, int cout, int L) {
    long i = (long)blockIdx.x * blockDim.x + threadIdx.x;
    long total = (long)16 * cout * L;
    if (i >= total) return;
    int o = (int)((i / L) % cout);
    float x = Y[i];
    x = (x - mean[o]) * rinv[o] * g[o] + be[o];
    Y[i] = geluf(x);
}

// ---------------------------------------------------------------------------
__global__ void k_maxpool(const float* __restrict__ X, float* __restrict__ Y,
                          int C, int S, int K) {
    long i = (long)blockIdx.x * blockDim.x + threadIdx.x;
    long total = (long)16 * C * S;
    if (i >= total) return;
    int b = (int)(i / ((long)C * S));
    int c = (int)((i / S) % C);
    int s = (int)(i % S);
    const float* p = X + ((size_t)b * C + c) * (size_t)S * K + (size_t)s * K;
    float mx = -1e30f;
    for (int k = 0; k < K; ++k) mx = fmaxf(mx, p[k]);
    Y[i] = mx;
}

__global__ void k_gmax(const float* __restrict__ X, float* __restrict__ Y, int C, int S) {
    int i = blockIdx.x * blockDim.x + threadIdx.x;
    if (i >= 16 * C) return;
    const float* p = X + (size_t)i * S;
    float mx = -1e30f;
    for (int s = 0; s < S; ++s) mx = fmaxf(mx, p[s]);
    Y[i] = mx;
}

// SA3 input: (B, 259, 128) = [l2_xyz^T ; l2]
__global__ void k_sa3cat(const float* __restrict__ nxyz2, const float* __restrict__ l2,
                         float* __restrict__ X) {
    long i = (long)blockIdx.x * blockDim.x + threadIdx.x;
    long total = (long)16 * 259 * 128;
    if (i >= total) return;
    int b = (int)(i / (259 * 128));
    int ch = (int)((i / 128) % 259);
    int s = (int)(i % 128);
    X[i] = (ch < 3) ? nxyz2[((size_t)b * 128 + s) * 3 + ch]
                    : l2[((size_t)b * 256 + (ch - 3)) * 128 + s];
}

// FP3 input: (B, 1280, 128) = [l2 ; broadcast(l3)]
__global__ void k_fp3cat(const float* __restrict__ l2, const float* __restrict__ l3,
                         float* __restrict__ X) {
    long i = (long)blockIdx.x * blockDim.x + threadIdx.x;
    long total = (long)16 * 1280 * 128;
    if (i >= total) return;
    int b = (int)(i / (1280 * 128));
    int ch = (int)((i / 128) % 1280);
    int s = (int)(i % 128);
    X[i] = (ch < 256) ? l2[((size_t)b * 256 + ch) * 128 + s]
                      : l3[(size_t)b * 1024 + (ch - 256)];
}

// 3-NN indices + normalized inverse-distance weights
__global__ void k_3nn(const float* __restrict__ xyz1, const float* __restrict__ xyz2,
                      int* __restrict__ nidx, float* __restrict__ nw, int N, int S) {
    int i = blockIdx.x * blockDim.x + threadIdx.x;
    if (i >= 16 * N) return;
    int b = i / N;
    const float* q = xyz1 + (size_t)i * 3;
    const float* P = xyz2 + (size_t)b * S * 3;
    float qx = q[0], qy = q[1], qz = q[2];
    float d0 = 1e30f, d1 = 1e30f, d2 = 1e30f;
    int i0 = 0, i1 = 0, i2 = 0;
    for (int j = 0; j < S; ++j) {
        float dx = qx - P[j * 3 + 0], dy = qy - P[j * 3 + 1], dz = qz - P[j * 3 + 2];
        float d = dx * dx + dy * dy + dz * dz;
        if (d < d0)      { d2 = d1; i2 = i1; d1 = d0; i1 = i0; d0 = d; i0 = j; }
        else if (d < d1) { d2 = d1; i2 = i1; d1 = d; i1 = j; }
        else if (d < d2) { d2 = d; i2 = j; }
    }
    float w0 = 1.0f / (d0 + 1e-8f), w1 = 1.0f / (d1 + 1e-8f), w2 = 1.0f / (d2 + 1e-8f);
    float ws = w0 + w1 + w2;
    nidx[(size_t)i * 3 + 0] = i0; nidx[(size_t)i * 3 + 1] = i1; nidx[(size_t)i * 3 + 2] = i2;
    nw[(size_t)i * 3 + 0] = w0 / ws; nw[(size_t)i * 3 + 1] = w1 / ws; nw[(size_t)i * 3 + 2] = w2 / ws;
}

// FP input: (B, C1+C2, N) = [points1 ; interp(points2)]
__global__ void k_fpcat(const float* __restrict__ p1, const float* __restrict__ p2,
                        const int* __restrict__ nidx, const float* __restrict__ nw,
                        float* __restrict__ X, int C1, int C2, int N, int S) {
    int C = C1 + C2;
    long i = (long)blockIdx.x * blockDim.x + threadIdx.x;
    long total = (long)16 * C * N;
    if (i >= total) return;
    int b  = (int)(i / ((long)C * N));
    int ch = (int)((i / N) % C);
    int n  = (int)(i % N);
    float v;
    if (ch < C1) {
        v = p1[((size_t)b * C1 + ch) * N + n];
    } else {
        int c2 = ch - C1;
        const int*   id = nidx + ((size_t)b * N + n) * 3;
        const float* w  = nw   + ((size_t)b * N + n) * 3;
        const float* pp = p2 + ((size_t)b * C2 + c2) * S;
        v = w[0] * pp[id[0]] + w[1] * pp[id[1]] + w[2] * pp[id[2]];
    }
    X[i] = v;
}

// ---------------------------------------------------------------------------
extern "C" void kernel_launch(void* const* d_in, const int* in_sizes, int n_in,
                              void* d_out, int out_size, void* d_ws, size_t ws_size,
                              hipStream_t stream) {
    (void)in_sizes; (void)n_in; (void)out_size; (void)ws_size;
    const int B = 16, N = 8192;

    auto F = [&](int i) { return (const float*)d_in[i]; };

    // Parameter unpack (dict insertion order, recursive)
    const float* x_t    = F(0);
    const float* t_in   = F(1);
    const float* time_W = F(2);
    const float* time_b = F(3);

    struct CC { const float *W, *b, *Wt, *bt, *g, *be; int cin, cout; };
    int cur = 4;
    auto getCC = [&](int cin, int cout) {
        CC p;
        p.W = F(cur++); p.b = F(cur++); p.Wt = F(cur++);
        p.bt = F(cur++); p.g = F(cur++); p.be = F(cur++);
        p.cin = cin; p.cout = cout;
        return p;
    };
    CC sa1[3] = { getCC(6, 64),    getCC(64, 64),   getCC(64, 128) };
    CC sa2[3] = { getCC(131, 128), getCC(128, 128), getCC(128, 256) };
    CC sa3[3] = { getCC(259, 256), getCC(256, 512), getCC(512, 1024) };
    CC fp3[2] = { getCC(1280, 256), getCC(256, 256) };
    CC fp2[2] = { getCC(384, 256),  getCC(256, 128) };
    CC fp1[3] = { getCC(131, 128),  getCC(128, 128), getCC(128, 128) };
    const float* head_W1 = F(cur++); const float* head_b1 = F(cur++);
    const float* head_g  = F(cur++); const float* head_be = F(cur++);
    const float* head_W2 = F(cur++); const float* head_b2 = F(cur++);

    // Workspace layout
    size_t off = 0;
    auto alloc = [&](size_t bytes) -> void* {
        void* p = (char*)d_ws + off;
        off += (bytes + 255) & ~(size_t)255;
        return p;
    };
    float* xyz   = (float*)alloc((size_t)B * N * 3 * 4);
    float* g2    = (float*)alloc((size_t)B * 128 * 4);
    float* tb    = (float*)alloc((size_t)B * 1280 * 4);
    float* ofs   = (float*)alloc((size_t)B * 1024 * 4);
    float* mean  = (float*)alloc(1024 * 4);
    float* rinv  = (float*)alloc(1024 * 4);
    int*   fps1  = (int*)alloc((size_t)B * 512 * 4);
    float* nxyz1 = (float*)alloc((size_t)B * 512 * 3 * 4);
    int*   ball1 = (int*)alloc((size_t)B * 512 * 32 * 4);
    int*   fps2  = (int*)alloc((size_t)B * 128 * 4);
    float* nxyz2 = (float*)alloc((size_t)B * 128 * 3 * 4);
    int*   ball2 = (int*)alloc((size_t)B * 128 * 64 * 4);
    float* l1    = (float*)alloc((size_t)B * 128 * 512 * 4);
    float* l2    = (float*)alloc((size_t)B * 256 * 128 * 4);
    float* l3    = (float*)alloc((size_t)B * 1024 * 4);
    int*   nidx2 = (int*)alloc((size_t)B * 512 * 3 * 4);
    float* nw2   = (float*)alloc((size_t)B * 512 * 3 * 4);
    int*   nidx1 = (int*)alloc((size_t)B * N * 3 * 4);
    float* nw1   = (float*)alloc((size_t)B * N * 3 * 4);
    float* PP0   = (float*)alloc((size_t)B * 128 * 16384 * 4); // 134 MB
    float* PP1   = (float*)alloc((size_t)B * 128 * 16384 * 4); // 134 MB

    float* A = PP0;   // current x_in / result
    float* Bq = PP1;  // scratch output
    auto swapPP = [&]() { float* tmp = A; A = Bq; Bq = tmp; };

    auto launch_gemm = [&](const float* Wm, const float* Xp, float* Yp,
                           int cout, int cin, int L) {
        dim3 g(L / 16, cdiv(cout, 32), B);
        if (cin % 4 == 0)
            k_gemm_t<true><<<g, 32, 0, stream>>>(Wm, Xp, ofs, Yp, cout, cin, L);
        else
            k_gemm_t<false><<<g, 32, 0, stream>>>(Wm, Xp, ofs, Yp, cout, cin, L);
    };

    auto conv_bn = [&](const float* Wm, const float* bias, const float* Wtm,
                       const float* btv, const float* gv, const float* bev,
                       int cin, int cout, int L, bool useTemb) {
        if (useTemb) {
            k_tb<<<cdiv((long)B * cin, 256), 256, 0, stream>>>(g2, Wtm, btv, tb, cin);
            k_ofs<<<cdiv((long)B * cout, 256), 256, 0, stream>>>(Wm, bias, tb, ofs, cin, cout);
        } else {
            k_biasofs<<<cdiv((long)B * cout, 256), 256, 0, stream>>>(bias, ofs, cout);
        }
        launch_gemm(Wm, A, Bq, cout, cin, L);
        k_bnstats<<<cout, 256, 0, stream>>>(Bq, mean, rinv, cout, L);
        k_bn_gelu<<<cdiv((long)B * cout * L, 256), 256, 0, stream>>>(Bq, mean, rinv, gv, bev, cout, L);
        swapPP();   // result now in A
    };
    auto cc = [&](const CC& p, int L) {
        conv_bn(p.W, p.b, p.Wt, p.bt, p.g, p.be, p.cin, p.cout, L, true);
    };

    // ---- time embedding + xyz transpose ----
    k_temb<<<1, 128, 0, stream>>>(t_in, time_W, time_b, g2);
    k_xpose<<<cdiv((long)B * N, 256), 256, 0, stream>>>(x_t, xyz, N);

    // ---- SA1: 512 centers, r=0.2, K=32, MLP [64,64,128] ----
    k_fps<<<B, 256, 0, stream>>>(xyz, fps1, N, 512);
    k_gather3<<<cdiv((long)B * 512, 256), 256, 0, stream>>>(xyz, fps1, nxyz1, N, 512);
    k_ball<<<cdiv((long)B * 512, 256), 256, 0, stream>>>(xyz, nxyz1, ball1, N, 512, 32, 0.04f);
    k_group<<<cdiv((long)B * 6 * 512 * 32, 256), 256, 0, stream>>>(xyz, nxyz1, x_t, ball1, Bq, N, 512, 32, 3);
    swapPP();
    for (int i = 0; i < 3; ++i) cc(sa1[i], 512 * 32);
    k_maxpool<<<cdiv((long)B * 128 * 512, 256), 256, 0, stream>>>(A, l1, 128, 512, 32);

    // ---- SA2: 128 centers, r=0.4, K=64, MLP [128,128,256] ----
    k_fps<<<B, 256, 0, stream>>>(nxyz1, fps2, 512, 128);
    k_gather3<<<cdiv((long)B * 128, 256), 256, 0, stream>>>(nxyz1, fps2, nxyz2, 512, 128);
    k_ball<<<cdiv((long)B * 128, 256), 256, 0, stream>>>(nxyz1, nxyz2, ball2, 512, 128, 64, 0.16f);
    k_group<<<cdiv((long)B * 131 * 128 * 64, 256), 256, 0, stream>>>(nxyz1, nxyz2, l1, ball2, Bq, 512, 128, 64, 128);
    swapPP();
    for (int i = 0; i < 3; ++i) cc(sa2[i], 128 * 64);
    k_maxpool<<<cdiv((long)B * 256 * 128, 256), 256, 0, stream>>>(A, l2, 256, 128, 64);

    // ---- SA3 (global): MLP [256,512,1024] ----
    k_sa3cat<<<cdiv((long)B * 259 * 128, 256), 256, 0, stream>>>(nxyz2, l2, Bq);
    swapPP();
    for (int i = 0; i < 3; ++i) cc(sa3[i], 128);
    k_gmax<<<cdiv((long)B * 1024, 256), 256, 0, stream>>>(A, l3, 1024, 128);

    // ---- FP3: broadcast l3, concat l2 -> MLP [256,256] ----
    k_fp3cat<<<cdiv((long)B * 1280 * 128, 256), 256, 0, stream>>>(l2, l3, Bq);
    swapPP();
    for (int i = 0; i < 2; ++i) cc(fp3[i], 128);
    // l2' (B,256,128) now in A

    // ---- FP2: 3NN(l1_xyz -> l2_xyz), concat l1 -> MLP [256,128] ----
    k_3nn<<<cdiv((long)B * 512, 256), 256, 0, stream>>>(nxyz1, nxyz2, nidx2, nw2, 512, 128);
    k_fpcat<<<cdiv((long)B * 384 * 512, 256), 256, 0, stream>>>(l1, A, nidx2, nw2, Bq, 128, 256, 512, 128);
    swapPP();
    for (int i = 0; i < 2; ++i) cc(fp2[i], 512);
    // l1' (B,128,512) now in A

    // ---- FP1: 3NN(xyz -> l1_xyz), concat x_t -> MLP [128,128,128] ----
    k_3nn<<<cdiv((long)B * N, 256), 256, 0, stream>>>(xyz, nxyz1, nidx1, nw1, N, 512);
    k_fpcat<<<cdiv((long)B * 131 * N, 256), 256, 0, stream>>>(x_t, A, nidx1, nw1, Bq, 3, 128, N, 512);
    swapPP();
    for (int i = 0; i < 3; ++i) cc(fp1[i], N);

    // ---- head: conv(128)+BN+GELU, then conv(3) -> d_out ----
    conv_bn(head_W1, head_b1, nullptr, nullptr, head_g, head_be, 128, 128, N, false);
    k_biasofs<<<cdiv((long)B * 3, 256), 256, 0, stream>>>(head_b2, ofs, 3);
    launch_gemm(head_W2, A, (float*)d_out, 3, 128, N);
}
